// GroupedQueryAttention_27625229648216
// MI455X (gfx1250) — matrix operand order repaired and checked by reading, not compiled, not run on hardware
//
#include <hip/hip_runtime.h>
#include <stdint.h>
#include <stddef.h>

// Problem constants (from reference)
#define B_    2
#define T_    2048
#define D_    2048
#define H_    16
#define KV_   4
#define HD_   128
#define KVD_  512   // D * KV / H
// scores scale = 1/sqrt(128), mixed with (1-ALIBI_W)=0.9
#define QK_SCALE (0.08838834764831845f * 0.9f)

typedef __bf16 bf16_t;
typedef __attribute__((ext_vector_type(8)))  bf16_t v8bf;
typedef __attribute__((ext_vector_type(16))) bf16_t v16bf;
typedef __attribute__((ext_vector_type(8)))  float  v8f;
typedef __attribute__((ext_vector_type(4)))  unsigned int u32x4;
typedef __attribute__((ext_vector_type(8)))  int i32x8;
typedef __attribute__((ext_vector_type(4)))  int i32x4;

static __device__ __forceinline__ bf16_t f2bf(float f) {
  union { float f; uint32_t u; } a; a.f = f;
  uint32_t r = a.u + 0x7FFFu + ((a.u >> 16) & 1u);  // round-to-nearest-even
  uint16_t h = (uint16_t)(r >> 16);
  bf16_t out;
  __builtin_memcpy(&out, &h, sizeof(out));
  return out;
}

static __device__ __forceinline__ v16bf cat16(v8bf lo, v8bf hi) {
  return __builtin_shufflevector(lo, hi, 0,1,2,3,4,5,6,7,8,9,10,11,12,13,14,15);
}

// ---------------------------------------------------------------------------
// f32 -> bf16 elementwise convert (for activations x)
// ---------------------------------------------------------------------------
__global__ void cvt_f32_bf16(const float* __restrict__ src,
                             bf16_t* __restrict__ dst, int n) {
  int i = blockIdx.x * blockDim.x + threadIdx.x;
  int stride = gridDim.x * blockDim.x;
  for (; i < n; i += stride) dst[i] = f2bf(src[i]);
}

// ---------------------------------------------------------------------------
// Weight convert + transpose: W[rows=K][cols=N] f32 -> Wt[N][K] bf16.
// One wave per 32x32 tile. Global->LDS leg uses the Tensor Data Mover
// (tensor_load_to_lds + s_wait_tensorcnt); 6-arg builtin on this toolchain.
// D# built per CDNA5 ISA (group0/group1 bitfields).
// ---------------------------------------------------------------------------
__global__ __launch_bounds__(32)
void wt_cvt_transpose(const float* __restrict__ src, bf16_t* __restrict__ dst,
                      int rows, int cols) {
  __shared__ alignas(16) float tile[32 * 32];
  const int lane = threadIdx.x;
  const int r0 = blockIdx.y * 32;   // k tile
  const int c0 = blockIdx.x * 32;   // n tile
#if __has_builtin(__builtin_amdgcn_tensor_load_to_lds) && __has_builtin(__builtin_amdgcn_s_wait_tensorcnt)
  {
    unsigned long long ga =
        (unsigned long long)(uintptr_t)(src + (size_t)r0 * cols + c0);
    unsigned int la = (unsigned int)(uintptr_t)(&tile[0]);  // LDS addr truncates
    // group0: count=1 | lds_addr | global_addr[56:0] | type=2 ("image")
    u32x4 g0 = { 1u, la, (unsigned int)ga,
                 (unsigned int)((ga >> 32) & 0x1FFFFFFull) | (2u << 30) };
    // group1: data_size=4B(code 2); tensor_dim0=cols, tensor_dim1=rows,
    //         tile_dim0=32, tile_dim1=32, tensor_dim0_stride=cols
    i32x8 g1 = { (2 << 16),
                 (int)((cols & 0xFFFF) << 16),
                 (int)(((cols >> 16) & 0xFFFF) | ((rows & 0xFFFF) << 16)),
                 (int)(((rows >> 16) & 0xFFFF) | (32 << 16)),
                 32,
                 cols,
                 0, 0 };
    i32x4 gz4 = { 0, 0, 0, 0 };
    i32x8 gz8 = { 0, 0, 0, 0, 0, 0, 0, 0 };
    __builtin_amdgcn_tensor_load_to_lds(g0, g1, gz4, gz4, gz8, 0);
    __builtin_amdgcn_s_wait_tensorcnt(0);
    asm volatile("" ::: "memory");
  }
#else
  for (int r = 0; r < 32; ++r)
    tile[r * 32 + lane] = src[(size_t)(r0 + r) * cols + c0 + lane];
  __syncthreads();   // single wave -> S_NOP
#endif
  // lane = local output row n; emit 32 contiguous k values as bf16
  alignas(16) bf16_t outv[32];
  #pragma unroll
  for (int kk = 0; kk < 32; ++kk) outv[kk] = f2bf(tile[kk * 32 + lane]);
  bf16_t* d = dst + (size_t)(c0 + lane) * rows + r0;
  #pragma unroll
  for (int c = 0; c < 4; ++c) *(v8bf*)(d + c * 8) = *(const v8bf*)(&outv[c * 8]);
}

// ---------------------------------------------------------------------------
// RoPE: src (B,T,heads,HD) f32 -> dst bf16 using freqs_cis (T,128)=[cos|sin]
// ---------------------------------------------------------------------------
__global__ void rope_bf16(const float* __restrict__ src,
                          const float* __restrict__ fc,
                          bf16_t* __restrict__ dst, int heads, int n) {
  int i = blockIdx.x * blockDim.x + threadIdx.x;
  int stride = gridDim.x * blockDim.x;
  for (; i < n; i += stride) {
    int d = i & (HD_ - 1);
    int t = (i / (HD_ * heads)) % T_;
    float x = src[i];
    float rot = (d < 64) ? -src[i + 64] : src[i - 64];
    float c = fc[t * HD_ + (d >> 1)];
    float s = fc[t * HD_ + 64 + (d >> 1)];
    dst[i] = f2bf(x * c + rot * s);
  }
}

// ---------------------------------------------------------------------------
// GEMM: C[M,N] = A[M,K] @ W[K,N] with W pre-transposed as Wt[N][K] bf16.
// No LDS, no barriers: both WMMA operands are per-lane contiguous global
// reads (weights are L2-resident: 192 MB global L2 >> 20 MB of weights).
// 256 threads = 8 waves; each wave owns a 32(M) x 64(N) tile = 8 accums,
// software-pipelined: next K-step operands are fetched before the current
// 8 WMMAs issue, so VMEM overlaps the XDL pipe instead of wait-0 serializing.
// mode: 0 = f32 row-major C, 1 = bf16 row-major C,
//       2 = bf16 transposed V store: vt[(b*KV+g)*HD+hd][T] from (row,col)
// ---------------------------------------------------------------------------
__global__ __launch_bounds__(256)
void gemm_bf16_wmma(const bf16_t* __restrict__ A, const bf16_t* __restrict__ Wt,
                    void* __restrict__ C, int M, int N, int K, int mode) {
  (void)M;
  const int lane = threadIdx.x & 31;
  const int wave = threadIdx.x >> 5;
  const int ln = lane & 15, hi = lane >> 4;
  const int mbase = blockIdx.y * 256 + wave * 32;
  const int nbase = blockIdx.x * 64;

  v8f z = {0,0,0,0,0,0,0,0};
  v8f acc[2][4];
  #pragma unroll
  for (int g = 0; g < 2; ++g)
    #pragma unroll
    for (int c = 0; c < 4; ++c) acc[g][c] = z;

  // A-operand rows (16-bit A layout): elems 0..7 = K(hi*8..), 8..15 = +16
  const bf16_t* arow0 = A + (size_t)(mbase + ln) * K + hi * 8;
  const bf16_t* arow1 = arow0 + (size_t)16 * K;
  // B-operand rows (16-bit B layout): lane ln = col, elems 0..7 = K(hi*8..), 8..15 = +16
  const bf16_t* brow = Wt + (size_t)(nbase + ln) * K + hi * 8;

  auto ldA = [&](const bf16_t* p, int k0) -> v16bf {
    return cat16(*(const v8bf*)(p + k0), *(const v8bf*)(p + k0 + 16));
  };
  auto ldB = [&](int c, int k0) -> v16bf {
    const bf16_t* bp = brow + (size_t)c * 16 * K + k0;
    return cat16(*(const v8bf*)bp, *(const v8bf*)(bp + 16));
  };

  // prologue: operands for k0 = 0
  v16bf av0 = ldA(arow0, 0), av1 = ldA(arow1, 0);
  v16bf bv[4];
  #pragma unroll
  for (int c = 0; c < 4; ++c) bv[c] = ldB(c, 0);

  for (int k0 = 0; k0 < K; k0 += 32) {
    // fetch next K-step operands first (wraps to 0 on the last iteration to
    // stay branchless; the extra loads are L0-hot and the results unused)
    const int kn = (k0 + 32 < K) ? (k0 + 32) : 0;
    v16bf av0n = ldA(arow0, kn);
    v16bf av1n = ldA(arow1, kn);
    v16bf bvn[4];
    #pragma unroll
    for (int c = 0; c < 4; ++c) bvn[c] = ldB(c, kn);

    // 8 WMMAs on the current operands while the next loads are in flight
    #pragma unroll
    for (int c = 0; c < 4; ++c) {
      acc[0][c] = __builtin_amdgcn_wmma_f32_16x16x32_bf16(false, av0, false, bv[c],
                                                          (short)0, acc[0][c], false, false);
      acc[1][c] = __builtin_amdgcn_wmma_f32_16x16x32_bf16(false, av1, false, bv[c],
                                                          (short)0, acc[1][c], false, false);
    }
    av0 = av0n; av1 = av1n;
    #pragma unroll
    for (int c = 0; c < 4; ++c) bv[c] = bvn[c];
  }

  #pragma unroll
  for (int g = 0; g < 2; ++g) {
    const int row0 = mbase + g * 16 + hi * 8;
    #pragma unroll
    for (int c = 0; c < 4; ++c) {
      const int col = nbase + c * 16 + ln;
      #pragma unroll
      for (int r = 0; r < 8; ++r) {
        const int row = row0 + r;
        if (mode == 0) {
          ((float*)C)[(size_t)row * N + col] = acc[g][c][r];
        } else if (mode == 1) {
          ((bf16_t*)C)[(size_t)row * N + col] = f2bf(acc[g][c][r]);
        } else {
          const int b = row >> 11, t = row & (T_ - 1);      // T_ = 2048
          const int gg = col >> 7, hd = col & (HD_ - 1);
          ((bf16_t*)C)[(((size_t)(b * KV_ + gg)) * HD_ + hd) * T_ + t] =
              f2bf(acc[g][c][r]);
        }
      }
    }
  }
}

// ---------------------------------------------------------------------------
// GQA flash attention, one wave per (b, head, 16-query tile).
// q: (B,T,H,HD) bf16 (RoPE'd)   k: (B,T,KV,HD) bf16   vt: (B,KV,HD,T) bf16
// scores = 0.9 * QK^T/sqrt(HD) - 0.1*slope*|m-n| + causal(-inf)
// K and V^T WMMA B-operands read straight from global (contiguous per lane);
// only P needs an LDS round-trip (C-layout -> A-layout).
// ---------------------------------------------------------------------------
__global__ __launch_bounds__(32)
void gqa_flash_wmma(const bf16_t* __restrict__ qm, const bf16_t* __restrict__ km,
                    const bf16_t* __restrict__ vt, bf16_t* __restrict__ om) {
  __shared__ alignas(16) bf16_t lP[16 * 32];    // P tile (m, k32), 1 KB
  const int lane = threadIdx.x & 31;
  const int ln = lane & 15, hi = lane >> 4;
  const int qt = blockIdx.x * 16;
  const int h  = blockIdx.y;
  const int b  = blockIdx.z;
  const int g  = h >> 2;                         // kv head (N_REP = 4)
  const float slope = exp2f(-8.0f * (float)(h + 1) / (float)H_);
  const float aw = 0.1f * slope;

  // Q in A-layout registers: 4 chunks of K=32 covering HD=128
  v16bf qa[4];
  const bf16_t* qrow = qm + ((size_t)(b * T_ + qt + ln)) * D_ + h * HD_;
  #pragma unroll
  for (int c = 0; c < 4; ++c) {
    int base = c * 32 + hi * 8;
    qa[c] = cat16(*(const v8bf*)(qrow + base), *(const v8bf*)(qrow + base + 16));
  }

  v8f z = {0,0,0,0,0,0,0,0};
  float mrow[8], lrow[8];
  v8f o_acc[8];
  #pragma unroll
  for (int r = 0; r < 8; ++r) { mrow[r] = -3.0e38f; lrow[r] = 0.0f; }
  #pragma unroll
  for (int sl = 0; sl < 8; ++sl) o_acc[sl] = z;

  const bf16_t* vbase = vt + ((size_t)(b * KV_ + g)) * HD_ * T_;

  const int nkb = (qt + 16 + 31) >> 5;   // causal: 32-key blocks up to diagonal
  for (int kb = 0; kb < nkb; ++kb) {
    const int k0 = kb * 32;
    if (kb + 1 < nkb) {    // global_prefetch_b8 of next K rows / V^T rows
      __builtin_prefetch(km + ((size_t)(b * T_ + k0 + 32 + lane)) * KVD_ + g * HD_, 0, 1);
      __builtin_prefetch(vbase + (size_t)(lane * 4) * T_ + k0 + 32, 0, 1);
    }

    // --- S = Q K^T for two 16-key tiles (K rows contiguous per lane)
    v8f st[2];
    #pragma unroll
    for (int t2 = 0; t2 < 2; ++t2) {
      st[t2] = z;
      const bf16_t* krow = km + ((size_t)(b * T_ + k0 + t2 * 16 + ln)) * KVD_ + g * HD_;
      #pragma unroll
      for (int c = 0; c < 4; ++c) {
        const bf16_t* kp = krow + c * 32 + hi * 8;
        v16bf bv = cat16(*(const v8bf*)kp, *(const v8bf*)(kp + 16));
        st[t2] = __builtin_amdgcn_wmma_f32_16x16x32_bf16(false, qa[c], false, bv,
                                                         (short)0, st[t2], false, false);
      }
      #pragma unroll
      for (int r = 0; r < 8; ++r) {
        int n = k0 + t2 * 16 + ln;
        int m = qt + hi * 8 + r;
        float sv = st[t2][r] * QK_SCALE - aw * fabsf((float)(m - n));
        st[t2][r] = (n > m) ? -3.0e38f : sv;
      }
    }

    // --- online softmax (rows live across 16 lanes at fixed vgpr index)
    float mt[8];
    #pragma unroll
    for (int r = 0; r < 8; ++r) mt[r] = fmaxf(st[0][r], st[1][r]);
    #pragma unroll
    for (int off = 8; off >= 1; off >>= 1) {
      #pragma unroll
      for (int r = 0; r < 8; ++r) mt[r] = fmaxf(mt[r], __shfl_xor(mt[r], off, 32));
    }
    float alpha[8];
    #pragma unroll
    for (int r = 0; r < 8; ++r) {
      float mn = fmaxf(mrow[r], mt[r]);
      alpha[r] = __expf(mrow[r] - mn);
      mrow[r] = mn;
    }
    float rs[8];
    #pragma unroll
    for (int r = 0; r < 8; ++r) {
      float p0 = __expf(st[0][r] - mrow[r]);
      float p1 = __expf(st[1][r] - mrow[r]);
      st[0][r] = p0; st[1][r] = p1;
      rs[r] = p0 + p1;
    }
    #pragma unroll
    for (int off = 8; off >= 1; off >>= 1) {
      #pragma unroll
      for (int r = 0; r < 8; ++r) rs[r] += __shfl_xor(rs[r], off, 32);
    }
    #pragma unroll
    for (int r = 0; r < 8; ++r) lrow[r] = lrow[r] * alpha[r] + rs[r];
    #pragma unroll
    for (int sl = 0; sl < 8; ++sl) {
      #pragma unroll
      for (int r = 0; r < 8; ++r) o_acc[sl][r] *= alpha[r];
    }

    // --- P (C-layout) -> LDS -> A-layout (same-wave LDS is in-order)
    #pragma unroll
    for (int t2 = 0; t2 < 2; ++t2) {
      #pragma unroll
      for (int r = 0; r < 8; ++r)
        lP[(hi * 8 + r) * 32 + t2 * 16 + ln] = f2bf(st[t2][r]);
    }
    const bf16_t* prow = lP + ln * 32 + hi * 8;
    v16bf pa = cat16(*(const v8bf*)prow, *(const v8bf*)(prow + 16));

    // --- O += P @ V : V^T rows give contiguous B-operand global reads
    #pragma unroll
    for (int sl = 0; sl < 8; ++sl) {
      const bf16_t* bp = vbase + (size_t)(sl * 16 + ln) * T_ + k0 + hi * 8;
      v16bf bv = cat16(*(const v8bf*)bp, *(const v8bf*)(bp + 16));
      o_acc[sl] = __builtin_amdgcn_wmma_f32_16x16x32_bf16(false, pa, false, bv,
                                                          (short)0, o_acc[sl], false, false);
    }
  }

  // epilogue: normalize by row sums, store bf16 (b, t, h, hd)
  bf16_t* orow = om + ((size_t)(b * T_ + qt + hi * 8)) * D_ + h * HD_ + ln;
  #pragma unroll
  for (int r = 0; r < 8; ++r) {
    float inv = 1.0f / lrow[r];
    #pragma unroll
    for (int sl = 0; sl < 8; ++sl)
      orow[(size_t)r * D_ + sl * 16] = f2bf(o_acc[sl][r] * inv);
  }
}

// ---------------------------------------------------------------------------
// host launcher
// ---------------------------------------------------------------------------
extern "C" void kernel_launch(void* const* d_in, const int* in_sizes, int n_in,
                              void* d_out, int out_size, void* d_ws, size_t ws_size,
                              hipStream_t stream) {
  (void)in_sizes; (void)n_in; (void)out_size; (void)ws_size;
  const float* x  = (const float*)d_in[0];
  // d_in[1] (mask) and d_in[3] (alibi_bias) are recomputed analytically
  const float* fc = (const float*)d_in[2];
  const float* Wq = (const float*)d_in[4];
  const float* Wk = (const float*)d_in[5];
  const float* Wv = (const float*)d_in[6];
  const float* Wo = (const float*)d_in[7];
  float* out = (float*)d_out;

  char* ws = (char*)d_ws;
  size_t off = 0;
  auto carve = [&](size_t bytes) -> void* {
    off = (off + 255) & ~(size_t)255;
    void* p = (void*)(ws + off);
    off += bytes;
    return p;
  };

  const size_t BT = (size_t)B_ * T_;           // 4096 rows
  bf16_t* xb  = (bf16_t*)carve(BT * D_ * 2);
  bf16_t* wqt = (bf16_t*)carve((size_t)D_ * D_ * 2);    // [N=2048][K=2048]
  bf16_t* wkt = (bf16_t*)carve((size_t)KVD_ * D_ * 2);  // [512][2048]
  bf16_t* wvt = (bf16_t*)carve((size_t)KVD_ * D_ * 2);
  bf16_t* wot = (bf16_t*)carve((size_t)D_ * D_ * 2);
  float*  qf  = (float*) carve(BT * D_ * 4);
  float*  kf  = (float*) carve(BT * KVD_ * 4);
  bf16_t* qrb = (bf16_t*)carve(BT * D_ * 2);
  bf16_t* krb = (bf16_t*)carve(BT * KVD_ * 2);
  bf16_t* vtb = (bf16_t*)carve((size_t)B_ * KV_ * HD_ * T_ * 2); // (B,KV,HD,T)
  bf16_t* ob  = (bf16_t*)carve(BT * D_ * 2);

  // 1) activation downconvert + weight convert/transpose (TDM global->LDS)
  cvt_f32_bf16<<<4096, 256, 0, stream>>>(x, xb, (int)(BT * D_));
  wt_cvt_transpose<<<dim3(D_ / 32,   D_ / 32), 32, 0, stream>>>(Wq, wqt, D_, D_);
  wt_cvt_transpose<<<dim3(KVD_ / 32, D_ / 32), 32, 0, stream>>>(Wk, wkt, D_, KVD_);
  wt_cvt_transpose<<<dim3(KVD_ / 32, D_ / 32), 32, 0, stream>>>(Wv, wvt, D_, KVD_);
  wt_cvt_transpose<<<dim3(D_ / 32,   D_ / 32), 32, 0, stream>>>(Wo, wot, D_, D_);

  // 2) projections (LDS-free, software-pipelined WMMA GEMMs)
  gemm_bf16_wmma<<<dim3(D_ / 64,   (unsigned)(BT / 256)), 256, 0, stream>>>(
      xb, wqt, qf, (int)BT, D_, D_, 0);
  gemm_bf16_wmma<<<dim3(KVD_ / 64, (unsigned)(BT / 256)), 256, 0, stream>>>(
      xb, wkt, kf, (int)BT, KVD_, D_, 0);
  gemm_bf16_wmma<<<dim3(KVD_ / 64, (unsigned)(BT / 256)), 256, 0, stream>>>(
      xb, wvt, vtb, (int)BT, KVD_, D_, 2);   // direct transposed V store

  // 3) RoPE in f32, emit bf16
  rope_bf16<<<4096, 256, 0, stream>>>(qf, fc, qrb, H_,  (int)(BT * D_));
  rope_bf16<<<1024, 256, 0, stream>>>(kf, fc, krb, KV_, (int)(BT * KVD_));

  // 4) flash attention (causal + ALiBi computed analytically)
  gqa_flash_wmma<<<dim3(T_ / 16, H_, B_), 32, 0, stream>>>(qrb, krb, vtb, ob);

  // 5) output projection -> f32 result
  gemm_bf16_wmma<<<dim3(D_ / 64, (unsigned)(BT / 256)), 256, 0, stream>>>(
      ob, wot, out, (int)BT, D_, D_, 0);
}
